// GCN_22213570855080
// MI455X (gfx1250) — compile-verified, matrix operand order checked by reading, not run on hardware
//
#include <hip/hip_runtime.h>

// ---------------------------------------------------------------------------
// GCN (2x GCNConv + linear head) for MI455X / gfx1250 (wave32, WMMA).
//
// Dense math:  v_wmma_f32_16x16x32_bf16, operands pre-packed as bf16 so each
//              16x32 fragment is two contiguous 16B chunks -> 2x b128 loads.
//              One wave per 16-row M-slab, A fragments register-resident
//              across all N tiles (K and NT compile-time -> fully unrolled).
// Aggregation: gather float4 from L2-resident features + global_atomic_add_f32
//              scatter (L2 RMW units). This is the roofline limiter (~1 GB of
//              L2 traffic vs ~1.3 GFLOP of dense math).
// ---------------------------------------------------------------------------

#define N_NODES 50000
#define N_EDGES 800000
#define IN_FEAT 64
#define HID 128
#define NCLS 32

typedef __attribute__((ext_vector_type(16))) __bf16 v16bf;
typedef __attribute__((ext_vector_type(8)))  float  v8f;

union Frag32B { uint4 u[2]; v16bf v; };     // one 16x32-slice bf16 fragment
union F32x8   { v8f   v; float e[8]; };
union BF4     { __bf16 e[4]; uint2 u; };    // 4 packed bf16 (8B store)

// Fragment load per ISA 7.12.2 (wave32, 16-bit A/B layout):
//   lane L: row/col = L%16, half = L/16
//   vector elems 0..7  hold K = k0 + half*8      .. +8   (16B contiguous)
//   vector elems 8..15 hold K = k0 + 16 + half*8 .. +8   (16B contiguous)
__device__ __forceinline__ v16bf load_frag(const __bf16* __restrict__ rowBase,
                                           int k0, int half)
{
    Frag32B f;
    f.u[0] = *(const uint4*)(rowBase + k0 + (half << 3));
    f.u[1] = *(const uint4*)(rowBase + k0 + 16 + (half << 3));
    return f.v;
}

// ---------------------------------------------------------------------------
// C[M x NT*16] = A[M x K] * Bt[NT*16 x K]^T  (+bias), all-bf16 operands.
// One wave per 16-row slab; A frags live in VGPRs across the NT loop.
// ---------------------------------------------------------------------------
template <int K, int NT>
__global__ __launch_bounds__(32)
void gemm_bf16_wmma(const __bf16* __restrict__ A,    // [M,K] row-major
                    const __bf16* __restrict__ Bt,   // [NT*16,K] (B transposed)
                    float* __restrict__ C, int ldc,
                    const float* __restrict__ bias)
{
    const int mBase = blockIdx.x << 4;
    const int lane  = threadIdx.x & 31;
    const int half  = lane >> 4;
    const int l16   = lane & 15;

    v16bf afrag[K / 32];
    const __bf16* Arow = A + (size_t)(mBase + l16) * K;
#pragma unroll
    for (int ks = 0; ks < K / 32; ++ks)
        afrag[ks] = load_frag(Arow, ks * 32, half);

#pragma unroll
    for (int nt = 0; nt < NT; ++nt) {
        const __bf16* Brow = Bt + (size_t)(nt * 16 + l16) * K;
        F32x8 acc;
        acc.v = (v8f){0.f, 0.f, 0.f, 0.f, 0.f, 0.f, 0.f, 0.f};
#pragma unroll
        for (int ks = 0; ks < K / 32; ++ks) {
            const v16bf bfrag = load_frag(Brow, ks * 32, half);
            acc.v = __builtin_amdgcn_wmma_f32_16x16x32_bf16(
                false, afrag[ks], false, bfrag, (short)0, acc.v, false, false);
        }
        const int   n  = nt * 16 + l16;
        const float bv = bias ? bias[n] : 0.0f;
#pragma unroll
        for (int r = 0; r < 8; ++r)
            C[(size_t)(mBase + r + (half << 3)) * ldc + n] = acc.e[r] + bv;
    }
}

// ---------------------------------------------------------------------------
// Operand packing: f32 -> bf16 (row-major copy / transposed weights)
// ---------------------------------------------------------------------------
__global__ void cvt_bf16_vec4(const float* __restrict__ in,
                              __bf16* __restrict__ out, int n4)
{
    int t = blockIdx.x * blockDim.x + threadIdx.x;
    if (t >= n4) return;
    const float4 v = *(const float4*)(in + (size_t)t * 4);
    BF4 o;
    o.e[0] = (__bf16)v.x; o.e[1] = (__bf16)v.y;
    o.e[2] = (__bf16)v.z; o.e[3] = (__bf16)v.w;
    *(uint2*)(out + (size_t)t * 4) = o.u;
}

__global__ void transpose_cvt_bf16(const float* __restrict__ W,  // [K,N]
                                   __bf16* __restrict__ Wt,      // [N,K]
                                   int K, int N)
{
    int t = blockIdx.x * blockDim.x + threadIdx.x;
    if (t >= K * N) return;
    const int k = t / N, n = t % N;
    Wt[(size_t)n * K + k] = (__bf16)W[t];
}

// ---------------------------------------------------------------------------
// Degree / normalization
// ---------------------------------------------------------------------------
__global__ void init_deg(float* __restrict__ deg)
{
    int i = blockIdx.x * blockDim.x + threadIdx.x;
    if (i < N_NODES) deg[i] = 1.0f;          // self-loop
}

__global__ void edge_deg(const int* __restrict__ dst, float* __restrict__ deg)
{
    int e = blockIdx.x * blockDim.x + threadIdx.x;
    if (e < N_EDGES) atomicAdd(&deg[dst[e]], 1.0f);
}

__global__ void calc_dinv(const float* __restrict__ deg, float* __restrict__ dinv)
{
    int i = blockIdx.x * blockDim.x + threadIdx.x;
    if (i < N_NODES) {
        const float d = deg[i];
        dinv[i] = d > 0.0f ? rsqrtf(d) : 0.0f;
    }
}

// ---------------------------------------------------------------------------
// Aggregation (F = 128 or 32): self-loop seed then per-edge atomic scatter.
// ---------------------------------------------------------------------------
template <int F>
__global__ void selfloop_seed(const float* __restrict__ h,
                              const float* __restrict__ dinv,
                              float* __restrict__ out)
{
    const int perNode = F / 4;
    int tid = blockIdx.x * blockDim.x + threadIdx.x;
    if (tid >= N_NODES * perNode) return;
    const int i = tid / perNode;
    const int f = (tid % perNode) << 2;
    float w = dinv[i]; w *= w;
    const float4 hv = *(const float4*)(h + (size_t)i * F + f);
    const float4 o = {hv.x * w, hv.y * w, hv.z * w, hv.w * w};
    *(float4*)(out + (size_t)i * F + f) = o;
}

template <int F>
__global__ void edge_scatter(const int* __restrict__ src,
                             const int* __restrict__ dst,
                             const float* __restrict__ dinv,
                             const float* __restrict__ h,
                             float* __restrict__ out)
{
    const int perEdge = F / 4;               // lanes per edge
    long long tid = (long long)blockIdx.x * blockDim.x + threadIdx.x;
    const int e = (int)(tid / perEdge);
    if (e >= N_EDGES) return;
    const int f = (int)(tid % perEdge) << 2;
    const int s = src[e], d = dst[e];
    const float nrm = dinv[s] * dinv[d];
    const float4 hv = *(const float4*)(h + (size_t)s * F + f);
    float* o = out + (size_t)d * F + f;
    atomicAdd(o + 0, hv.x * nrm);
    atomicAdd(o + 1, hv.y * nrm);
    atomicAdd(o + 2, hv.z * nrm);
    atomicAdd(o + 3, hv.w * nrm);
}

// ---------------------------------------------------------------------------
// Epilogues.
// Layer 1: h1_bf = bf16(relu(hagg1 + b1))       (only bf16 copy is consumed)
// Layer 2: h2 (f32, returned) += b2, plus bf16 copy for the head GEMM
// ---------------------------------------------------------------------------
__global__ void bias_relu_to_bf16(const float* __restrict__ hin,
                                  const float* __restrict__ bias,
                                  __bf16* __restrict__ hout)
{
    int tid = blockIdx.x * blockDim.x + threadIdx.x;      // N * 32 threads
    if (tid >= N_NODES * (HID / 4)) return;
    const int f = (tid & (HID / 4 - 1)) << 2;
    float4 v = *(const float4*)(hin + (size_t)tid * 4);
    v.x = fmaxf(v.x + bias[f + 0], 0.f);
    v.y = fmaxf(v.y + bias[f + 1], 0.f);
    v.z = fmaxf(v.z + bias[f + 2], 0.f);
    v.w = fmaxf(v.w + bias[f + 3], 0.f);
    BF4 o;
    o.e[0] = (__bf16)v.x; o.e[1] = (__bf16)v.y;
    o.e[2] = (__bf16)v.z; o.e[3] = (__bf16)v.w;
    *(uint2*)(hout + (size_t)tid * 4) = o.u;
}

__global__ void bias_f32_and_bf16(float* __restrict__ h,
                                  const float* __restrict__ bias,
                                  __bf16* __restrict__ hbf)
{
    int tid = blockIdx.x * blockDim.x + threadIdx.x;      // N * 8 threads
    if (tid >= N_NODES * (NCLS / 4)) return;
    const int f = (tid & (NCLS / 4 - 1)) << 2;
    float4 v = *(float4*)(h + (size_t)tid * 4);
    v.x += bias[f + 0]; v.y += bias[f + 1];
    v.z += bias[f + 2]; v.w += bias[f + 3];
    *(float4*)(h + (size_t)tid * 4) = v;
    BF4 o;
    o.e[0] = (__bf16)v.x; o.e[1] = (__bf16)v.y;
    o.e[2] = (__bf16)v.z; o.e[3] = (__bf16)v.w;
    *(uint2*)(hbf + (size_t)tid * 4) = o.u;
}

// ---------------------------------------------------------------------------
// Launch
// ---------------------------------------------------------------------------
extern "C" void kernel_launch(void* const* d_in, const int* in_sizes, int n_in,
                              void* d_out, int out_size, void* d_ws, size_t ws_size,
                              hipStream_t stream)
{
    const float* x  = (const float*)d_in[0];
    const int*   ei = (const int*)  d_in[1];
    const float* W1 = (const float*)d_in[2];
    const float* b1 = (const float*)d_in[3];
    const float* W2 = (const float*)d_in[4];
    const float* b2 = (const float*)d_in[5];
    const float* Wc = (const float*)d_in[6];
    const float* bc = (const float*)d_in[7];

    const int* src = ei;             // edge_index[0]
    const int* dst = ei + N_EDGES;   // edge_index[1]

    // ---- workspace carve-up (256B aligned) --------------------------------
    char* p = (char*)d_ws;
    auto alloc = [&](size_t bytes) {
        char* r = p;
        p += (bytes + 255) & ~(size_t)255;
        return r;
    };
    float*  deg   = (float*) alloc((size_t)N_NODES * 4);
    float*  dinv  = (float*) alloc((size_t)N_NODES * 4);
    __bf16* x_bf  = (__bf16*)alloc((size_t)N_NODES * IN_FEAT * 2);
    __bf16* W1t   = (__bf16*)alloc((size_t)IN_FEAT * HID * 2);   // [HID, IN_FEAT]
    __bf16* W2t   = (__bf16*)alloc((size_t)HID * NCLS * 2);      // [NCLS, HID]
    __bf16* Wct   = (__bf16*)alloc((size_t)NCLS * NCLS * 2);     // [NCLS, NCLS]
    float*  hagg1 = (float*) alloc((size_t)N_NODES * HID * 4);
    char*   region= alloc((size_t)N_NODES * HID * 4);            // aliased below

    float*  hw1   = (float*)region;                                    // gemm1 out
    __bf16* h1_bf = (__bf16*)region;                                   // after hw1 dead
    float*  hw2   = (float*)(region + (size_t)N_NODES * HID * 2);      // gemm2 out
    __bf16* h2_bf = (__bf16*)(region + (size_t)N_NODES * HID * 2
                                     + (size_t)N_NODES * NCLS * 4);    // head input

    float* out = (float*)d_out;                      // [N, 32] logits
    float* h2  = out + (size_t)N_NODES * NCLS;       // [N, 32] returned h

    const int TB = 256;

    // ---- operand packing --------------------------------------------------
    cvt_bf16_vec4<<<(N_NODES * IN_FEAT / 4 + TB - 1) / TB, TB, 0, stream>>>(
        x, x_bf, N_NODES * IN_FEAT / 4);
    transpose_cvt_bf16<<<(IN_FEAT * HID + TB - 1) / TB, TB, 0, stream>>>(
        W1, W1t, IN_FEAT, HID);
    transpose_cvt_bf16<<<(HID * NCLS + TB - 1) / TB, TB, 0, stream>>>(
        W2, W2t, HID, NCLS);
    transpose_cvt_bf16<<<(NCLS * NCLS + TB - 1) / TB, TB, 0, stream>>>(
        Wc, Wct, NCLS, NCLS);

    // ---- normalization ----------------------------------------------------
    init_deg <<<(N_NODES + TB - 1) / TB, TB, 0, stream>>>(deg);
    edge_deg <<<(N_EDGES + TB - 1) / TB, TB, 0, stream>>>(dst, deg);
    calc_dinv<<<(N_NODES + TB - 1) / TB, TB, 0, stream>>>(deg, dinv);

    // ---- layer 1: hw1 = x @ W1 -------------------------------------------
    gemm_bf16_wmma<IN_FEAT, HID / 16><<<N_NODES / 16, 32, 0, stream>>>(
        x_bf, W1t, hw1, HID, nullptr);

    selfloop_seed<HID><<<(N_NODES * (HID / 4) + TB - 1) / TB, TB, 0, stream>>>(
        hw1, dinv, hagg1);
    edge_scatter<HID><<<((long long)N_EDGES * (HID / 4) + TB - 1) / TB, TB, 0, stream>>>(
        src, dst, dinv, hw1, hagg1);
    bias_relu_to_bf16<<<(N_NODES * (HID / 4) + TB - 1) / TB, TB, 0, stream>>>(
        hagg1, b1, h1_bf);

    // ---- layer 2: hw2 = relu(h1) @ W2 ------------------------------------
    gemm_bf16_wmma<HID, NCLS / 16><<<N_NODES / 16, 32, 0, stream>>>(
        h1_bf, W2t, hw2, NCLS, nullptr);

    selfloop_seed<NCLS><<<(N_NODES * (NCLS / 4) + TB - 1) / TB, TB, 0, stream>>>(
        hw2, dinv, h2);
    edge_scatter<NCLS><<<((long long)N_EDGES * (NCLS / 4) + TB - 1) / TB, TB, 0, stream>>>(
        src, dst, dinv, hw2, h2);
    bias_f32_and_bf16<<<(N_NODES * (NCLS / 4) + TB - 1) / TB, TB, 0, stream>>>(
        h2, b2, h2_bf);

    // ---- head: out = h2 @ Wc + bc ----------------------------------------
    gemm_bf16_wmma<NCLS, NCLS / 16><<<N_NODES / 16, 32, 0, stream>>>(
        h2_bf, Wct, out, NCLS, bc);
}
